// CosineSimConv2D_33277406609724
// MI455X (gfx1250) — matrix operand order repaired and checked by reading, not compiled, not run on hardware
//
#include <hip/hip_runtime.h>

typedef __attribute__((ext_vector_type(16))) _Float16 v16h;
typedef __attribute__((ext_vector_type(8)))  float    v8f;
typedef __attribute__((ext_vector_type(4)))  int      v4i;

#define AS1 __attribute__((address_space(1)))
#define AS3 __attribute__((address_space(3)))

#define B_   8
#define E_   128
#define HH   112
#define WW   112
#define O_   256
#define D_   1152            // E_ * 9
#define EPSN 1e-12f
#define NSTEP 36             // 9 taps * 4 channel chunks

#if defined(__has_builtin)
#if __has_builtin(__builtin_amdgcn_global_load_async_to_lds_b128)
#define HAVE_ASYNC_LDS 1
#endif
#if __has_builtin(__builtin_amdgcn_s_wait_asynccnt)
#define HAVE_WAIT_ASYNC 1
#endif
#endif

__device__ __forceinline__ void wait_asynccnt0() {
#if defined(HAVE_WAIT_ASYNC)
  __builtin_amdgcn_s_wait_asynccnt(0);
#else
  asm volatile("s_wait_asynccnt 0x0" ::: "memory");
#endif
}

// ---------------------------------------------------------------------------
// Kernel 1: normalize each weight row, convert to f16, write pre-swizzled into
// WMMA B-fragment order: frag index f = (tap*4 + cchunk)*16 + otile,
// within fragment: lane = n + 16*(kk>>4), half index = kk & 15.
// ---------------------------------------------------------------------------
__global__ void wprep_kernel(const float* __restrict__ w, _Float16* __restrict__ wfrag) {
  const int o = blockIdx.x;   // 0..255
  const int t = threadIdx.x;  // 0..255
  __shared__ float red[256];
  float s = 0.0f;
  for (int d = t; d < D_; d += 256) { float v = w[o * D_ + d]; s += v * v; }
  red[t] = s;
  __syncthreads();
  for (int st = 128; st > 0; st >>= 1) {
    if (t < st) red[t] += red[t + st];
    __syncthreads();
  }
  const float inv = 1.0f / fmaxf(sqrtf(red[0]), EPSN);
  for (int d = t; d < D_; d += 256) {
    const float v = w[o * D_ + d] * inv;
    const int tap = d % 9;
    const int c   = d / 9;
    const int cc  = c >> 5;
    const int kk  = c & 31;
    const int ot  = o >> 4;
    const int n   = o & 15;
    const int lane = n + 16 * (kk >> 4);
    const int hidx = kk & 15;
    const int f = (tap * 4 + cc) * 16 + ot;
    wfrag[(size_t)f * 512 + lane * 16 + hidx] = (_Float16)v;
  }
}

// ---------------------------------------------------------------------------
// Kernel 2: per-pixel sum over channels of x^2  (coalesced across w).
// ---------------------------------------------------------------------------
__global__ void sq_kernel(const float* __restrict__ x, float* __restrict__ ssum) {
  const int idx = blockIdx.x * 256 + threadIdx.x;   // 0 .. B*H*W-1
  const int hw  = idx % (HH * WW);
  const int b   = idx / (HH * WW);
  const float* p = x + (size_t)b * E_ * HH * WW + hw;
  float s = 0.0f;
  for (int c = 0; c < E_; ++c) {
    const float v = p[(size_t)c * HH * WW];
    s += v * v;
  }
  ssum[idx] = s;
}

// ---------------------------------------------------------------------------
// Kernel 3: 3x3 zero-padded box sum of squared sums -> inverse patch norm.
// ---------------------------------------------------------------------------
__global__ void boxinv_kernel(const float* __restrict__ ssum, float* __restrict__ invn) {
  const int idx = blockIdx.x * 256 + threadIdx.x;
  const int w = idx % WW;
  const int h = (idx / WW) % HH;
  const int b = idx / (HH * WW);
  const float* p = ssum + (size_t)b * HH * WW;
  float s = 0.0f;
  for (int dh = -1; dh <= 1; ++dh)
    for (int dw = -1; dw <= 1; ++dw) {
      const int hh = h + dh, ww = w + dw;
      if (hh >= 0 && hh < HH && ww >= 0 && ww < WW) s += p[hh * WW + ww];
    }
  invn[idx] = 1.0f / fmaxf(sqrtf(s), EPSN);
}

// ---------------------------------------------------------------------------
// Kernel 4: implicit-im2col GEMM with v_wmma_f32_16x16x32_f16, double-buffered
// LDS pipeline, one barrier per K-step. Per step: LDS fragment loads first,
// then staging for step k+1 (async B copy + A pack/store), then 8 WMMAs from
// registers -> DS loads precede DS stores in the queue, so WMMAs only wait on
// the loads and the staging drains under the matrix math.
// ---------------------------------------------------------------------------
__global__ __launch_bounds__(256, 1)
void gemm_kernel(const float* __restrict__ x,
                 const _Float16* __restrict__ wfrag,
                 const float* __restrict__ invn,
                 float* __restrict__ out) {
  __shared__ __align__(32) _Float16 Afrag[2][4][32][16];    // 8 KB
  __shared__ __align__(32) _Float16 Bfrag[2][16][32][16];   // 32 KB

  const int tid   = threadIdx.x;
  const int lane  = tid & 31;
  const int wid   = tid >> 5;
  const int r     = wid >> 1;   // m-tile (h row) 0..3
  const int ohalf = wid & 1;    // which 128 outputs
  const int b  = blockIdx.z;
  const int h0 = blockIdx.y * 4;
  const int w0 = blockIdx.x * 16;

  // A-staging thread mapping: channel PAIR (even,odd share one frag dword),
  // row, 4-wide w segment. Consecutive threads read contiguous w.
  const int kp   = tid >> 4;        // 0..15 -> channels 2kp, 2kp+1
  const int rm   = tid & 15;
  const int sr   = rm >> 2;         // row 0..3
  const int mseg = rm & 3;          // 4-w segment 0..3
  const int kk0  = kp * 2;
  // 16-bit A fragment layout (matches B): lanes 0-15 K=0-15; lanes 16-31 K=16-31
  const int asel = kk0 >> 4;
  const int adw  = (kk0 & 15) >> 1;   // dword index 0..7

  auto stage = [&](int step, int buf) {
    const int tap = step >> 2;      // 0..8
    const int cc  = step & 3;       // 0..3
    const int dh  = tap / 3 - 1;
    const int dw  = tap % 3 - 1;

    // ---- B: contiguous 16 KB fragment-ordered copy, global -> LDS ----
    const uint4* gsrc = (const uint4*)(wfrag + (size_t)((tap * 4 + cc) * 16) * 512);
    uint4* bdst = (uint4*)(&Bfrag[buf][0][0][0]);
#if defined(HAVE_ASYNC_LDS)
#pragma unroll
    for (int i = 0; i < 4; ++i) {
      __builtin_amdgcn_global_load_async_to_lds_b128(
          (AS1 v4i*)(gsrc + tid + i * 256),
          (AS3 v4i*)(bdst + tid + i * 256),
          /*offset=*/0, /*cpol=*/0);
    }
#else
    {
      uint4 t0 = gsrc[tid];
      uint4 t1 = gsrc[tid + 256];
      uint4 t2 = gsrc[tid + 512];
      uint4 t3 = gsrc[tid + 768];
      bdst[tid]       = t0;
      bdst[tid + 256] = t1;
      bdst[tid + 512] = t2;
      bdst[tid + 768] = t3;
    }
#endif

    // ---- A: 2 channels x 1 row x 4 w per thread, packed b32 LDS stores ----
    const int hh = h0 + sr + dh;
    const bool hok = (hh >= 0) && (hh < HH);
    const float* p0 =
        x + (((size_t)b * E_ + cc * 32 + kk0) * HH + (hok ? hh : 0)) * WW;
    const float* p1 = p0 + (size_t)HH * WW;
    const int wstart = w0 + dw + mseg * 4;
    float v0[4], v1[4];
    if (hok && wstart >= 0 && (wstart + 3) < WW) {
#pragma unroll
      for (int j = 0; j < 4; ++j) { v0[j] = p0[wstart + j]; v1[j] = p1[wstart + j]; }
    } else {
#pragma unroll
      for (int j = 0; j < 4; ++j) {
        const int ww_ = wstart + j;
        const bool ok = hok && ww_ >= 0 && ww_ < WW;
        v0[j] = ok ? p0[ww_] : 0.0f;
        v1[j] = ok ? p1[ww_] : 0.0f;
      }
    }
    unsigned int* adst = (unsigned int*)&Afrag[buf][sr][0][0];  // [32 lanes][8 dw]
#pragma unroll
    for (int j = 0; j < 4; ++j) {
      const int m = mseg * 4 + j;
      union { _Float16 h[2]; unsigned int u; } pk;
      pk.h[0] = (_Float16)v0[j];
      pk.h[1] = (_Float16)v1[j];
      adst[(m + 16 * asel) * 8 + adw] = pk.u;
    }
  };

  v8f acc[8] = {};

  stage(0, 0);
  for (int step = 0; step < NSTEP; ++step) {
    const int cur = step & 1;
#if defined(HAVE_ASYNC_LDS)
    wait_asynccnt0();
#endif
    __syncthreads();                 // buf[cur] staged; buf[cur^1] fully read

    // fragment loads FIRST (so DS loads precede next-step DS stores)
    const v16h a = *(const v16h*)(&Afrag[cur][r][lane][0]);
    v16h bm[8];
#pragma unroll
    for (int ot = 0; ot < 8; ++ot)
      bm[ot] = *(const v16h*)(&Bfrag[cur][ohalf * 8 + ot][lane][0]);

    if (step + 1 < NSTEP) stage(step + 1, cur ^ 1);

#pragma unroll
    for (int ot = 0; ot < 8; ++ot) {
      acc[ot] = __builtin_amdgcn_wmma_f32_16x16x32_f16(
          /*neg_a=*/false, a, /*neg_b=*/false, bm[ot],
          /*c_mod=*/(short)0, acc[ot], /*reuse_a=*/false, /*reuse_b=*/false);
    }
  }

  // ---- epilogue: scale by inverse patch norm, store (contiguous in w) ----
  // D layout: lane holds N = lane&15, rows M = 8*(lane>>4) + v  (v = vgpr idx)
  const int mrow = 8 * (lane >> 4);
  const int n    = lane & 15;
  const float* invp = invn + ((size_t)b * HH + (h0 + r)) * WW + w0 + mrow;
  float sc[8];
#pragma unroll
  for (int v = 0; v < 8; ++v) sc[v] = invp[v];

#pragma unroll
  for (int ot = 0; ot < 8; ++ot) {
    const int o = ohalf * 128 + ot * 16 + n;
    float* op = out + (((size_t)b * O_ + o) * HH + (h0 + r)) * WW + w0 + mrow;
    float4 lo = make_float4(acc[ot][0] * sc[0], acc[ot][1] * sc[1],
                            acc[ot][2] * sc[2], acc[ot][3] * sc[3]);
    float4 hi = make_float4(acc[ot][4] * sc[4], acc[ot][5] * sc[5],
                            acc[ot][6] * sc[6], acc[ot][7] * sc[7]);
    *(float4*)(op)     = lo;
    *(float4*)(op + 4) = hi;
  }
}

// ---------------------------------------------------------------------------
extern "C" void kernel_launch(void* const* d_in, const int* in_sizes, int n_in,
                              void* d_out, int out_size, void* d_ws, size_t ws_size,
                              hipStream_t stream) {
  const float* x = (const float*)d_in[0];   // (8,128,112,112)
  const float* w = (const float*)d_in[1];   // (256,1152)
  float* out = (float*)d_out;               // (8,256,112,112)

  char* ws = (char*)d_ws;
  _Float16* wfrag = (_Float16*)ws;                               // 589824 B
  float* ssum = (float*)(ws + 589824);                           // 401408 B
  float* invn = (float*)(ws + 589824 + 401408);                  // 401408 B

  wprep_kernel<<<O_, 256, 0, stream>>>(w, wfrag);
  sq_kernel<<<(B_ * HH * WW) / 256, 256, 0, stream>>>(x, ssum);
  boxinv_kernel<<<(B_ * HH * WW) / 256, 256, 0, stream>>>(ssum, invn);

  dim3 grid(WW / 16, HH / 4, B_);
  gemm_kernel<<<grid, 256, 0, stream>>>(x, wfrag, invn, out);
}
